// GeometricAttentionLayer_197568496154
// MI455X (gfx1250) — compile-verified, matrix operand order checked by reading, not run on hardware
//
#include <hip/hip_runtime.h>
#include <hip/hip_bf16.h>

typedef __attribute__((ext_vector_type(16))) _Float16 v16h;
typedef __attribute__((ext_vector_type(8)))  _Float16 h8;
typedef __attribute__((ext_vector_type(8)))  float    v8f;

constexpr int GAc = 16;     // multivector components
constexpr int Ec  = 256;    // embedding channels
constexpr int Sc  = 2048;   // sequence length
constexpr int Bc  = 4;      // batch
constexpr int NC  = 256;    // channels (= H*D)
constexpr int LDP = 264;    // padded row stride (halves) for g-major LDS tiles
constexpr int TOKENS = Bc * Sc;           // 8192
constexpr unsigned IPMASK = 0x871Du;      // blades {0,2,3,4,8,9,10,15}

// K index inside a 32-wide K slice for fragment element j, per A/B layout tables
__device__ __forceinline__ int kmap(int j, int lanehi) {
    return (j & 7) + ((j >= 8) ? 16 : 0) + (lanehi ? 8 : 0);
}

// Preload one weight matrix's A-fragments for this wave's 16-row tile into VGPRs.
// A[m,k] = W[row, ks*32+k]; lane holds row m = lane&15, K split by lane half.
__device__ __forceinline__ void load_w_frags(const float* __restrict__ W, int row,
                                             int lanehi, v16h* a) {
#pragma unroll
    for (int ks = 0; ks < 8; ++ks) {
        v16h f;
#pragma unroll
        for (int j = 0; j < 16; ++j) {
            f[j] = (_Float16)W[row * Ec + ks * 32 + kmap(j, lanehi)];
        }
        a[ks] = f;
    }
}

// B-fragment (K=32 x N=16) from a g-major LDS tile: B[k,n] = tile[g=n][e=ks*32+k].
// Two aligned 16B LDS loads per fragment.
__device__ __forceinline__ v16h load_b_frag(const _Float16* tile, int n, int ks, int lanehi) {
    const int base = n * LDP + ks * 32 + (lanehi ? 8 : 0);
    h8 lo = *(const h8*)(tile + base);
    h8 hi = *(const h8*)(tile + base + 16);
    v16h b;
#pragma unroll
    for (int j = 0; j < 8; ++j) { b[j] = lo[j]; b[j + 8] = hi[j]; }
    return b;
}

// ---------------------------------------------------------------------------
// Kernel 1: q/k equi-linear projections (WMMA) + blade inner product -> scores
// scores laid out [b][c][s] (transposed) for the sequence softmax.
// ---------------------------------------------------------------------------
__global__ __launch_bounds__(512) void ga_qk_scores(
    const float* __restrict__ Q, const float* __restrict__ K,
    const float* __restrict__ Wq, const float* __restrict__ Wk,
    float* __restrict__ scores, int tok_per_wg)
{
    __shared__ __attribute__((aligned(16))) _Float16 qt[GAc * LDP];
    __shared__ __attribute__((aligned(16))) _Float16 kt[GAc * LDP];

    const int tid    = threadIdx.x;
    const int wave   = tid >> 5;        // 0..15 -> output-channel tile
    const int lane   = tid & 31;
    const int lanehi = lane >> 4;
    const int n      = lane & 15;       // B column (= g), also A row within tile
    const int row    = wave * 16 + n;

    v16h aq[8], ak[8];
    load_w_frags(Wq, row, lanehi, aq);
    load_w_frags(Wk, row, lanehi, ak);

    const float gm = ((IPMASK >> n) & 1u) ? 1.0f : 0.0f;  // blade mask for column g=n
    const int tok0 = blockIdx.x * tok_per_wg;

    for (int t = 0; t < tok_per_wg; ++t) {
        const int tok = tok0 + t;
        const float* xq = Q + (size_t)tok * (Ec * GAc);
        const float* xk = K + (size_t)tok * (Ec * GAc);

        // Stage token tiles f32 -> f16, transposed to g-major LDS (coalesced reads).
        for (int i = tid; i < Ec * GAc; i += 512) {
            const int e = i >> 4, g = i & 15;
            qt[g * LDP + e] = (_Float16)xq[i];
            kt[g * LDP + e] = (_Float16)xk[i];
        }
        if (t + 1 < tok_per_wg) {   // global_prefetch_b8 for the next token tile
            __builtin_prefetch(xq + Ec * GAc + tid * 8, 0, 0);
            __builtin_prefetch(xk + Ec * GAc + tid * 8, 0, 0);
        }
        __syncthreads();

        v8f cq = {}, ck = {};
#pragma unroll
        for (int ks = 0; ks < 8; ++ks) {
            const v16h bq = load_b_frag(qt, n, ks, lanehi);
            const v16h bk = load_b_frag(kt, n, ks, lanehi);
            cq = __builtin_amdgcn_wmma_f32_16x16x32_f16(false, aq[ks], false, bq,
                                                        (short)0, cq, false, false);
            ck = __builtin_amdgcn_wmma_f32_16x16x32_f16(false, ak[ks], false, bk,
                                                        (short)0, ck, false, false);
        }

        // iprod[c] = sum_{g in IP8} q[c,g]*k[c,g]; reduce over the 16-lane half.
        float pr[8];
#pragma unroll
        for (int r = 0; r < 8; ++r) {
            float p = cq[r] * ck[r] * gm;
            p += __shfl_xor(p, 1, 32);
            p += __shfl_xor(p, 2, 32);
            p += __shfl_xor(p, 4, 32);
            p += __shfl_xor(p, 8, 32);
            pr[r] = p;
        }
        if (n == 0) {
            const int bi = tok / Sc, si = tok - bi * Sc;
#pragma unroll
            for (int r = 0; r < 8; ++r) {
                const int c = wave * 16 + (lanehi ? r + 8 : r);
                scores[((size_t)bi * NC + c) * Sc + si] = pr[r] * (1.0f / 16.0f);
            }
        }
        __syncthreads();
    }
}

// ---------------------------------------------------------------------------
// Kernel 2: softmax over the sequence axis, one block per (b,c) row, in place.
// ---------------------------------------------------------------------------
__global__ __launch_bounds__(256) void ga_softmax(float* __restrict__ sc)
{
    float* p = sc + (size_t)blockIdx.x * Sc;
    const int tid = threadIdx.x;
    const int w = tid >> 5, ln = tid & 31;
    __shared__ float red[8];

    float x[8];
    float m = -3.4e38f;
#pragma unroll
    for (int r = 0; r < 8; ++r) { x[r] = p[r * 256 + tid]; m = fmaxf(m, x[r]); }
    for (int off = 1; off < 32; off <<= 1) m = fmaxf(m, __shfl_xor(m, off, 32));
    if (ln == 0) red[w] = m;
    __syncthreads();
    if (tid == 0) { float mm = red[0]; for (int i = 1; i < 8; ++i) mm = fmaxf(mm, red[i]); red[0] = mm; }
    __syncthreads();
    m = red[0];

    float s = 0.f;
#pragma unroll
    for (int r = 0; r < 8; ++r) { x[r] = __expf(x[r] - m); s += x[r]; }
    for (int off = 1; off < 32; off <<= 1) s += __shfl_xor(s, off, 32);
    __syncthreads();
    if (ln == 0) red[w] = s;
    __syncthreads();
    if (tid == 0) { float ss = red[0]; for (int i = 1; i < 8; ++i) ss += red[i]; red[0] = ss; }
    __syncthreads();
    const float inv = 1.0f / red[0];
#pragma unroll
    for (int r = 0; r < 8; ++r) p[r * 256 + tid] = x[r] * inv;
}

// ---------------------------------------------------------------------------
// Kernel 3: fused  out = Wo @ (attn * (Wv @ X_v))  per token, both via WMMA.
// ---------------------------------------------------------------------------
__global__ __launch_bounds__(512) void ga_v_out(
    const float* __restrict__ V, const float* __restrict__ Wv,
    const float* __restrict__ Wo, const float* __restrict__ attn,
    float* __restrict__ out, int tok_per_wg)
{
    __shared__ __attribute__((aligned(16))) _Float16 vt[GAc * LDP];  // staged V tile
    __shared__ __attribute__((aligned(16))) _Float16 vp[GAc * LDP];  // scaled Wv@X tile

    const int tid    = threadIdx.x;
    const int wave   = tid >> 5;
    const int lane   = tid & 31;
    const int lanehi = lane >> 4;
    const int n      = lane & 15;
    const int row    = wave * 16 + n;

    v16h av[8], ao[8];
    load_w_frags(Wv, row, lanehi, av);
    load_w_frags(Wo, row, lanehi, ao);

    const int tok0 = blockIdx.x * tok_per_wg;

    for (int t = 0; t < tok_per_wg; ++t) {
        const int tok = tok0 + t;
        const int bi = tok / Sc, si = tok - bi * Sc;
        const float* xv = V + (size_t)tok * (Ec * GAc);

        for (int i = tid; i < Ec * GAc; i += 512) {
            const int e = i >> 4, g = i & 15;
            vt[g * LDP + e] = (_Float16)xv[i];
        }
        if (t + 1 < tok_per_wg)
            __builtin_prefetch(xv + Ec * GAc + tid * 8, 0, 0);
        __syncthreads();

        // Phase 1: Vp = Wv @ X   (16 row-tile per wave)
        v8f cv = {};
#pragma unroll
        for (int ks = 0; ks < 8; ++ks) {
            const v16h bv = load_b_frag(vt, n, ks, lanehi);
            cv = __builtin_amdgcn_wmma_f32_16x16x32_f16(false, av[ks], false, bv,
                                                        (short)0, cv, false, false);
        }
        // Scale each channel row by its attention weight, deposit g-major f16.
#pragma unroll
        for (int r = 0; r < 8; ++r) {
            const int c = wave * 16 + (lanehi ? r + 8 : r);
            const float a = attn[((size_t)bi * NC + c) * Sc + si];
            vp[n * LDP + c] = (_Float16)(cv[r] * a);
        }
        __syncthreads();

        // Phase 2: Y = Wo @ Vp
        v8f cy = {};
#pragma unroll
        for (int ks = 0; ks < 8; ++ks) {
            const v16h by = load_b_frag(vp, n, ks, lanehi);
            cy = __builtin_amdgcn_wmma_f32_16x16x32_f16(false, ao[ks], false, by,
                                                        (short)0, cy, false, false);
        }
        float* o = out + (size_t)tok * (Ec * GAc);
#pragma unroll
        for (int r = 0; r < 8; ++r) {
            const int oc = wave * 16 + (lanehi ? r + 8 : r);
            o[oc * GAc + n] = cy[r];
        }
        __syncthreads();
    }
}

// ---------------------------------------------------------------------------
extern "C" void kernel_launch(void* const* d_in, const int* in_sizes, int n_in,
                              void* d_out, int out_size, void* d_ws, size_t ws_size,
                              hipStream_t stream) {
    (void)in_sizes; (void)n_in; (void)out_size; (void)ws_size;
    const float* Q  = (const float*)d_in[0];
    const float* K  = (const float*)d_in[1];
    const float* V  = (const float*)d_in[2];
    const float* Wq = (const float*)d_in[3];
    const float* Wk = (const float*)d_in[4];
    const float* Wv = (const float*)d_in[5];
    const float* Wo = (const float*)d_in[6];
    float* out    = (float*)d_out;
    float* scores = (float*)d_ws;   // [Bc][NC][Sc] = 8 MB scratch

    constexpr int TOK_PER_WG = 32;
    constexpr int NWG = TOKENS / TOK_PER_WG;   // 256 workgroups

    ga_qk_scores<<<NWG, 512, 0, stream>>>(Q, K, Wq, Wk, scores, TOK_PER_WG);
    ga_softmax<<<Bc * NC, 256, 0, stream>>>(scores);
    ga_v_out<<<NWG, 512, 0, stream>>>(V, Wv, Wo, scores, out, TOK_PER_WG);
}
